// CCVPEDecoder_42769284333638
// MI455X (gfx1250) — matrix operand order repaired and checked by reading, not compiled, not run on hardware
//
#include <hip/hip_runtime.h>
#include <math.h>

// ===========================================================================
// CCVPE decoder for MI455X (gfx1250, wave32, WMMA).
// Heavy GEMMs (Linear + all 3x3 convs) run on v_wmma_f32_16x16x32_f16 with
// f32 accumulation. Block tile = 128(M) x 32(N): 8 waves each own a 16-row
// M-subtile and two 16x16 accumulators; the 32-col B (weight) tile is staged
// once per block cooperatively (float4 loads), A is staged per-wave with
// branchless clamped loads (no exec-mask ladders, batched global loads).
// Bandwidth-bound stages (deconv2x2, scores, l2norm, softmax, concats) stay
// on plain VALU kernels — they are HBM-limited at 23.3 TB/s regardless.
// ===========================================================================

typedef __attribute__((ext_vector_type(16))) _Float16 v16h;
typedef __attribute__((ext_vector_type(8)))  float    v8f;

#define TPB 256

enum { MODE_PLAIN = 0, MODE_CONV3 = 1 };

// ---------------------------------------------------------------------------
// C[M,N] = A[M,K] * Bt[N,K]^T + bias[N]  (opt. ReLU), NCHW store:
//   row m -> (b, oh, ow), out[((b*Ctot + ch0 + n)*H + oh)*W + ow]
// MODE_PLAIN : A dense row-major [M,K] f32
// MODE_CONV3 : implicit im2col of NCHW img [B,Cin,H,W], 3x3 SAME (pad 1),
//              K = Cin*9 matching OIHW weight flattening.
// Requirements guaranteed by this network: M % 128 == 0, K % 4 == 0, W >= 16.
// ---------------------------------------------------------------------------
template<int MODE>
__global__ __launch_bounds__(256)
void gemm_wmma(const float* __restrict__ A, const float* __restrict__ Bt,
               const float* __restrict__ bias, float* __restrict__ out,
               int M, int N, int K, int Cin, int H, int W,
               int Ctot, int ch0, int relu)
{
    __shared__ _Float16 sA[8][16][32];   // per-wave A tile  [wave][m][k]
    __shared__ _Float16 sB[2][16][32];   // block-shared B   [ntile][n][k]

    const int lane = threadIdx.x & 31;
    const int wv   = threadIdx.x >> 5;
    const int n0   = blockIdx.x * 32;
    const int m0   = (blockIdx.y * 8 + wv) * 16;
    const int HWp  = H * W;

    // rows m0..m0+15 share (b, oh); ow = ow0 + t  (W multiple of 16)
    int bB, oh, ow0;
    {
        bB = m0 / HWp;
        int p = m0 - bB * HWp;
        oh  = p / W;
        ow0 = p - oh * W;
    }

    // cooperative-B staging coordinates (fixed per thread)
    const int btile = threadIdx.x >> 7;          // 0..1
    const int bg    = threadIdx.x & 127;
    const int bnn   = bg & 15;
    const int bkg   = bg >> 4;                   // 0..7 (4 k's each)
    const int bcol  = n0 + btile * 16 + bnn;
    const int bcolc = (bcol < N) ? bcol : (N - 1);

    v8f acc0 = {}, acc1 = {};

    for (int kc = 0; kc < K; kc += 32) {
        // ---- stage A tile (16x32): lane owns column kk = lane, rows = t ----
        {
            int k    = kc + lane;
            int kok  = (k < K);
            int kcl  = kok ? k : (K - 1);
            float vals[16];
            if (MODE == MODE_PLAIN) {
                const float* colp = A + (long)m0 * K + kcl;
                #pragma unroll
                for (int t = 0; t < 16; ++t)
                    vals[t] = colp[(long)t * K];
                #pragma unroll
                for (int t = 0; t < 16; ++t)
                    sA[wv][t][lane] = (_Float16)(kok ? vals[t] : 0.f);
            } else {
                int ci = kcl / 9;
                int r  = kcl - ci * 9;
                int kh = r / 3;
                int kw = r - kh * 3;
                int ih = oh + kh - 1;
                int ihOK = kok && (ih >= 0) && (ih < H);
                int ihc  = ih < 0 ? 0 : (ih >= H ? H - 1 : ih);
                const float* rowp = A + (((long)bB * Cin + ci) * H + ihc) * W;
                int iwb = ow0 + kw - 1;
                #pragma unroll
                for (int t = 0; t < 16; ++t) {
                    int iw  = iwb + t;
                    int iwc = iw < 0 ? 0 : (iw >= W ? W - 1 : iw);
                    vals[t] = rowp[iwc];
                }
                #pragma unroll
                for (int t = 0; t < 16; ++t) {
                    int iw = iwb + t;
                    int ok = ihOK && (iw >= 0) && (iw < W);
                    sA[wv][t][lane] = (_Float16)(ok ? vals[t] : 0.f);
                }
            }
        }
        // ---- stage B tiles (2 x 16x32) cooperatively: 1 float4 per thread --
        {
            int k   = kc + (bkg << 2);
            int ok  = (bcol < N) && (k < K);            // K%4==0 -> k+3<K too
            int kcl = (k < K) ? k : (K - 4);
            float4 v = *(const float4*)(Bt + (long)bcolc * K + kcl);
            _Float16* dst = &sB[btile][bnn][bkg << 2];
            dst[0] = (_Float16)(ok ? v.x : 0.f);
            dst[1] = (_Float16)(ok ? v.y : 0.f);
            dst[2] = (_Float16)(ok ? v.z : 0.f);
            dst[3] = (_Float16)(ok ? v.w : 0.f);
        }
        __syncthreads();
        // ---- fragments per ISA 7.12.2 (wave32, 16-bit) ----
        // A: lane L -> row M=L&15, halves K={kh..kh+7, kh+16..kh+23}, kh=(L>>4)*8
        // B: lane L -> col N=L&15, halves K={kh..kh+7, kh+16..kh+23}, kh=(L>>4)*8
        {
            int lm  = lane & 15;
            int khA = (lane >> 4) << 3;
            int kbB = (lane >> 4) << 4;
            v16h af, bf0, bf1;
            #pragma unroll
            for (int j = 0; j < 8; ++j) {
                af[j]     = sA[wv][lm][khA + j];
                af[j + 8] = sA[wv][lm][khA + 16 + j];
            }
            #pragma unroll
            for (int j = 0; j < 8; ++j) {
                bf0[j]     = sB[0][lm][khA + j];
                bf0[j + 8] = sB[0][lm][khA + 16 + j];
                bf1[j]     = sB[1][lm][khA + j];
                bf1[j + 8] = sB[1][lm][khA + 16 + j];
            }
            acc0 = __builtin_amdgcn_wmma_f32_16x16x32_f16(
                       false, af, false, bf0, (short)0, acc0, false, false);
            acc1 = __builtin_amdgcn_wmma_f32_16x16x32_f16(
                       false, af, false, bf1, (short)0, acc1, false, false);
        }
        __syncthreads();
    }

    // ---- epilogue: VGPR r holds M=(lane>>4)*8+r, N=lane&15 ----
    int nn = lane & 15;
    int mb = (lane >> 4) << 3;
    long obase = (((long)bB * Ctot + ch0) * H + oh) * W + (ow0 + mb);
    #pragma unroll
    for (int t2 = 0; t2 < 2; ++t2) {
        int col = n0 + t2 * 16 + nn;
        if (col < N) {
            float bv = bias[col];
            v8f a = t2 ? acc1 : acc0;
            #pragma unroll
            for (int r = 0; r < 8; ++r) {
                float v = a[r] + bv;
                if (relu) v = fmaxf(v, 0.f);
                out[obase + (long)col * HWp + r] = v;
            }
        }
    }
}

// ---------------------------------------------------------------------------
// map_volume [B,1280,16,16] -> packed A [512, 5120]
// ---------------------------------------------------------------------------
__global__ void pack_chunks_kernel(const float* __restrict__ mv,
                                   float* __restrict__ A)
{
    long idx = (long)blockIdx.x * blockDim.x + threadIdx.x;
    if (idx >= 512L * 5120L) return;
    int col = (int)(idx % 5120);
    int row = (int)(idx / 5120);
    int b = row >> 6, g = row & 63, gy = g >> 3, gx = g & 7;
    int c = col >> 2, r = col & 3, h = r >> 1, w = r & 1;
    A[idx] = mv[((long)b * 1280 + c) * 256 + (2 * gy + h) * 16 + (2 * gx + w)];
}

// ---------------------------------------------------------------------------
// ConvTranspose2d k=2 s=2 into concat buffer channels [0, Co)
// ---------------------------------------------------------------------------
__global__ void deconv2_kernel(const float* __restrict__ x,
                               const float* __restrict__ w,
                               const float* __restrict__ bias,
                               float* __restrict__ out,
                               int Bv, int Ci, int Co, int H, int W, int Ctot)
{
    long idx = (long)blockIdx.x * blockDim.x + threadIdx.x;
    long H2 = 2L * H, W2 = 2L * W;
    long total = (long)Bv * Co * H2 * W2;
    if (idx >= total) return;
    int ox = (int)(idx % W2); long t = idx / W2;
    int oy = (int)(t % H2);   t /= H2;
    int o  = (int)(t % Co);
    int b  = (int)(t / Co);
    int h = oy >> 1, p = oy & 1, ww = ox >> 1, q = ox & 1;
    float acc = bias[o];
    const float* xp = x + ((long)b * Ci * H + h) * W + ww;
    const float* wp = w + ((long)o * 4) + p * 2 + q;
    for (int i = 0; i < Ci; ++i)
        acc += xp[(long)i * H * W] * wp[(long)i * Co * 4];
    out[(((long)b * Ctot + o) * H2 + oy) * W2 + ox] = acc;
}

// ---------------------------------------------------------------------------
// Rotation-correlation scores (20 hypotheses, circular roll over channels)
// ---------------------------------------------------------------------------
__global__ void scores_kernel(const float* __restrict__ x,
                              const float* __restrict__ g,
                              float* __restrict__ sc,
                              int Bv, int C, int L, int shift, long HW)
{
    long idx = (long)blockIdx.x * blockDim.x + threadIdx.x;
    long total = (long)Bv * 20 * HW;
    if (idx >= total) return;
    long hw = idx % HW; long t = idx / HW;
    int i = (int)(t % 20);
    int b = (int)(t / 20);
    int base = i * shift;
    float gn = 0.f, sn = 0.f, s = 0.f;
    for (int c = 0; c < L; ++c) {
        float gv = g[((long)b * L + c) * HW + hw];
        int xc = base + c; if (xc >= C) xc -= C;           // single wrap suffices
        float xv = x[((long)b * C + xc) * HW + hw];
        gn += gv * gv; sn += xv * xv; s += gv * xv;
    }
    sc[((long)b * 20 + i) * HW + hw] = s / (sqrtf(sn) * sqrtf(gn));
}

// ---------------------------------------------------------------------------
// xcat[b] = concat( max_i sc[b,i], l2norm(x[b]) ) -> [B, C+1, H, W]
// ---------------------------------------------------------------------------
__global__ void xcat_kernel(const float* __restrict__ x,
                            const float* __restrict__ sc,
                            float* __restrict__ xc,
                            int Bv, int C, long HW)
{
    long idx = (long)blockIdx.x * blockDim.x + threadIdx.x;
    if (idx >= (long)Bv * HW) return;
    long hw = idx % HW; int b = (int)(idx / HW);
    float mx = -3.4e38f;
    for (int i = 0; i < 20; ++i)
        mx = fmaxf(mx, sc[((long)b * 20 + i) * HW + hw]);
    float ss = 0.f;
    for (int c = 0; c < C; ++c) {
        float v = x[((long)b * C + c) * HW + hw];
        ss += v * v;
    }
    float inv = 1.f / fmaxf(sqrtf(ss), 1e-12f);
    xc[((long)b * (C + 1)) * HW + hw] = mx;
    for (int c = 0; c < C; ++c)
        xc[((long)b * (C + 1) + 1 + c) * HW + hw] =
            x[((long)b * C + c) * HW + hw] * inv;
}

// ---------------------------------------------------------------------------
__global__ void copy_block_kernel(const float* __restrict__ blk,
                                  float* __restrict__ dst,
                                  int Bv, int Cb, int ch0, int Ctot, long HW)
{
    long idx = (long)blockIdx.x * blockDim.x + threadIdx.x;
    long total = (long)Bv * Cb * HW;
    if (idx >= total) return;
    long hw = idx % HW; long t = idx / HW;
    int c = (int)(t % Cb);
    int b = (int)(t / Cb);
    dst[((long)b * Ctot + ch0 + c) * HW + hw] = blk[idx];
}

// ---------------------------------------------------------------------------
// xo0 = concat( ms0[B,20,8,8], l2norm(fdm[B,1280,8,8]) ) -> [B,1300,8,8]
// ---------------------------------------------------------------------------
__global__ void build_xo0_kernel(const float* __restrict__ fdm,
                                 const float* __restrict__ ms0,
                                 float* __restrict__ xo, int Bv)
{
    long idx = (long)blockIdx.x * blockDim.x + threadIdx.x;
    if (idx >= (long)Bv * 64) return;
    long hw = idx & 63; int b = (int)(idx >> 6);
    for (int i = 0; i < 20; ++i)
        xo[((long)b * 1300 + i) * 64 + hw] = ms0[((long)b * 20 + i) * 64 + hw];
    float ss = 0.f;
    for (int c = 0; c < 1280; ++c) {
        float v = fdm[((long)b * 1280 + c) * 64 + hw];
        ss += v * v;
    }
    float inv = 1.f / fmaxf(sqrtf(ss), 1e-12f);
    for (int c = 0; c < 1280; ++c)
        xo[((long)b * 1300 + 20 + c) * 64 + hw] =
            fdm[((long)b * 1280 + c) * 64 + hw] * inv;
}

// ---------------------------------------------------------------------------
__global__ void softmax_kernel(const float* __restrict__ logits,
                               float* __restrict__ hm, int n)
{
    __shared__ float red[TPB];
    int b = blockIdx.x;
    const float* p = logits + (long)b * n;
    float* o = hm + (long)b * n;
    float mx = -3.4e38f;
    for (int i = threadIdx.x; i < n; i += TPB) mx = fmaxf(mx, p[i]);
    red[threadIdx.x] = mx; __syncthreads();
    for (int s = TPB / 2; s > 0; s >>= 1) {
        if (threadIdx.x < s)
            red[threadIdx.x] = fmaxf(red[threadIdx.x], red[threadIdx.x + s]);
        __syncthreads();
    }
    mx = red[0]; __syncthreads();
    float sm = 0.f;
    for (int i = threadIdx.x; i < n; i += TPB) sm += __expf(p[i] - mx);
    red[threadIdx.x] = sm; __syncthreads();
    for (int s = TPB / 2; s > 0; s >>= 1) {
        if (threadIdx.x < s) red[threadIdx.x] += red[threadIdx.x + s];
        __syncthreads();
    }
    float inv = 1.f / red[0];
    for (int i = threadIdx.x; i < n; i += TPB) o[i] = __expf(p[i] - mx) * inv;
}

// ---------------------------------------------------------------------------
__global__ void l2norm2_kernel(const float* __restrict__ x,
                               float* __restrict__ o, int Bv, long HW)
{
    long idx = (long)blockIdx.x * blockDim.x + threadIdx.x;
    if (idx >= (long)Bv * HW) return;
    long hw = idx % HW; int b = (int)(idx / HW);
    float a = x[((long)b * 2 + 0) * HW + hw];
    float c = x[((long)b * 2 + 1) * HW + hw];
    float inv = 1.f / fmaxf(sqrtf(a * a + c * c), 1e-12f);
    o[((long)b * 2 + 0) * HW + hw] = a * inv;
    o[((long)b * 2 + 1) * HW + hw] = c * inv;
}

// ===========================================================================
// Host orchestration
// ===========================================================================
static const int kGRD_LENS[6]      = {320, 160, 80, 40, 20, 10};
static const int kLOC_DECONV[6][2] = {{1281,1024},{641,320},{321,160},{161,80},{81,40},{41,16}};
static const int kLOC_CONV[6][3]   = {{1344,640,640},{432,320,320},{200,160,160},{104,80,80},{56,40,40},{16,16,1}};
static const int kORI_DECONV[6][2] = {{1300,1024},{640,256},{256,128},{128,64},{64,32},{32,16}};
static const int kORI_CONV[6][3]   = {{1344,640,640},{368,256,256},{168,128,128},{88,64,64},{48,32,32},{16,16,2}};
static const int kBLOCK_C[5]       = {320, 112, 40, 24, 16};
static const int kHWX[6]           = {8, 16, 32, 64, 128, 256}; // spatial at level entry
static const int BATCH = 8;

extern "C" void kernel_launch(void* const* d_in, const int* in_sizes, int n_in,
                              void* d_out, int out_size, void* d_ws, size_t ws_size,
                              hipStream_t stream)
{
    (void)in_sizes; (void)n_in; (void)out_size; (void)ws_size;

    // ---- input map (setup_inputs dict order; params flattened in dict order)
    // 0-5 grd_map0..5 | 6-11 grd_desc (unused) | 12-16 block0..4 | 17 map_volume
    // 18.. loc_deconv (w,b)x6 | 30.. loc_conv (w1,b1,w2,b2)x6
    // 54.. ori_deconv (w,b)x6 | 66.. ori_conv (w1,b1,w2,b2)x6 | 90 lin_w | 91 lin_b
    const float* grd_map[6]; for (int l = 0; l < 6; ++l) grd_map[l] = (const float*)d_in[l];
    const float* blocks[5];  for (int i = 0; i < 5; ++i) blocks[i]  = (const float*)d_in[12 + i];
    const float* map_volume = (const float*)d_in[17];
    const float *ldw[6], *ldb[6], *lc1w[6], *lc1b[6], *lc2w[6], *lc2b[6];
    const float *odw[6], *odb[6], *oc1w[6], *oc1b[6], *oc2w[6], *oc2b[6];
    for (int l = 0; l < 6; ++l) {
        ldw[l]  = (const float*)d_in[18 + 2 * l];  ldb[l]  = (const float*)d_in[19 + 2 * l];
        lc1w[l] = (const float*)d_in[30 + 4 * l];  lc1b[l] = (const float*)d_in[31 + 4 * l];
        lc2w[l] = (const float*)d_in[32 + 4 * l];  lc2b[l] = (const float*)d_in[33 + 4 * l];
        odw[l]  = (const float*)d_in[54 + 2 * l];  odb[l]  = (const float*)d_in[55 + 2 * l];
        oc1w[l] = (const float*)d_in[66 + 4 * l];  oc1b[l] = (const float*)d_in[67 + 4 * l];
        oc2w[l] = (const float*)d_in[68 + 4 * l];  oc2b[l] = (const float*)d_in[69 + 4 * l];
    }
    const float* lin_w = (const float*)d_in[90];
    const float* lin_b = (const float*)d_in[91];

    // ---- d_out layout (floats): logits | heatmap | xo | ms0..ms5
    float* outf = (float*)d_out;
    const long nLog = (long)BATCH * 262144;
    float* o_logits = outf;
    float* o_heat   = o_logits + nLog;
    float* o_xo     = o_heat + nLog;
    float* o_ms[6];
    {
        long off = 0;
        float* base = o_xo + (long)BATCH * 2 * 262144;
        for (int l = 0; l < 6; ++l) {
            o_ms[l] = base + off;
            long hw = (long)kHWX[l] * kHWX[l];
            off += (long)BATCH * 20 * hw;
        }
    }

    // ---- workspace bump allocator (256B aligned) ----
    char* wsb = (char*)d_ws;
    size_t cur = 0;
    auto alloc = [&](size_t bytes) -> float* {
        size_t o = (cur + 255) & ~(size_t)255;
        cur = o + bytes;
        return (float*)(wsb + o);
    };
    float* Apack = alloc(512L * 5120 * 4);
    float* fdm   = alloc((long)BATCH * 1280 * 64 * 4);
    float* xo0   = alloc((long)BATCH * 1300 * 64 * 4);
    const size_t PBYTES = (size_t)BATCH * 40 * 65536 * 4; // max persistent level out
    float* P0 = alloc(PBYTES);
    float* P1 = alloc(PBYTES);
    size_t transient_base = cur;

    auto blocks1d = [](long total) { return (unsigned)((total + TPB - 1) / TPB); };
    auto launch_gemm = [&](int mode, const float* A, const float* Bt, const float* bias,
                           float* out, int M, int N, int K, int Cin, int H, int W,
                           int Ctot, int ch0, int relu) {
        dim3 g((N + 31) / 32, M / 128);   // M % 128 == 0 for all GEMMs here
        if (mode == MODE_PLAIN)
            gemm_wmma<MODE_PLAIN><<<g, 256, 0, stream>>>(A, Bt, bias, out, M, N, K,
                                                         Cin, H, W, Ctot, ch0, relu);
        else
            gemm_wmma<MODE_CONV3><<<g, 256, 0, stream>>>(A, Bt, bias, out, M, N, K,
                                                         Cin, H, W, Ctot, ch0, relu);
    };

    // ================= Phase 0: per-chunk Linear -> fdm [B,1280,8,8] ========
    pack_chunks_kernel<<<blocks1d(512L * 5120), TPB, 0, stream>>>(map_volume, Apack);
    launch_gemm(MODE_PLAIN, Apack, lin_w, lin_b, fdm,
                512, 1280, 5120, /*Cin*/0, 8, 8, /*Ctot*/1280, 0, 0);

    // ================= Localization path ====================================
    const float* x = fdm;
    int Cx = 1280;
    for (int l = 0; l < 6; ++l) {
        cur = transient_base;                     // reset transient arena
        int H = kHWX[l];  long HW = (long)H * H;
        int L = kGRD_LENS[l], shift = 64 >> l;
        scores_kernel<<<blocks1d((long)BATCH * 20 * HW), TPB, 0, stream>>>(
            x, grd_map[l], o_ms[l], BATCH, Cx, L, shift, HW);
        float* xcat = alloc((long)BATCH * (Cx + 1) * HW * 4);
        xcat_kernel<<<blocks1d((long)BATCH * HW), TPB, 0, stream>>>(
            x, o_ms[l], xcat, BATCH, Cx, HW);
        int Ci = kLOC_DECONV[l][0], Co = kLOC_DECONV[l][1];
        int Ccat = kLOC_CONV[l][0];
        int H2 = 2 * H;  long HW2 = (long)H2 * H2;
        float* cat = alloc((long)BATCH * Ccat * HW2 * 4);
        deconv2_kernel<<<blocks1d((long)BATCH * Co * HW2), TPB, 0, stream>>>(
            xcat, ldw[l], ldb[l], cat, BATCH, Ci, Co, H, H, Ccat);
        if (l < 5)
            copy_block_kernel<<<blocks1d((long)BATCH * kBLOCK_C[l] * HW2), TPB, 0, stream>>>(
                blocks[l], cat, BATCH, kBLOCK_C[l], Co, Ccat, HW2);
        int Cm = kLOC_CONV[l][1], CoC = kLOC_CONV[l][2];
        float* mid = alloc((long)BATCH * Cm * HW2 * 4);
        launch_gemm(MODE_CONV3, cat, lc1w[l], lc1b[l], mid,
                    BATCH * (int)HW2, Cm, Ccat * 9, Ccat, H2, H2, Cm, 0, 1);
        float* out = (l == 5) ? o_logits : ((l & 1) ? P1 : P0);
        launch_gemm(MODE_CONV3, mid, lc2w[l], lc2b[l], out,
                    BATCH * (int)HW2, CoC, Cm * 9, Cm, H2, H2, CoC, 0, 0);
        x = out; Cx = CoC;
    }
    softmax_kernel<<<BATCH, TPB, 0, stream>>>(o_logits, o_heat, 262144);

    // ================= Orientation path =====================================
    build_xo0_kernel<<<blocks1d((long)BATCH * 64), TPB, 0, stream>>>(fdm, o_ms[0], xo0, BATCH);
    const float* xo = xo0;
    float* oriFinal = nullptr;
    for (int l = 0; l < 6; ++l) {
        cur = transient_base;
        int H = kHWX[l];
        int Ci = kORI_DECONV[l][0], Co = kORI_DECONV[l][1];
        int Ccat = kORI_CONV[l][0];
        int H2 = 2 * H;  long HW2 = (long)H2 * H2;
        float* cat = alloc((long)BATCH * Ccat * HW2 * 4);
        deconv2_kernel<<<blocks1d((long)BATCH * Co * HW2), TPB, 0, stream>>>(
            xo, odw[l], odb[l], cat, BATCH, Ci, Co, H, H, Ccat);
        if (l < 5)
            copy_block_kernel<<<blocks1d((long)BATCH * kBLOCK_C[l] * HW2), TPB, 0, stream>>>(
                blocks[l], cat, BATCH, kBLOCK_C[l], Co, Ccat, HW2);
        int Cm = kORI_CONV[l][1], CoC = kORI_CONV[l][2];
        float* mid = alloc((long)BATCH * Cm * HW2 * 4);
        launch_gemm(MODE_CONV3, cat, oc1w[l], oc1b[l], mid,
                    BATCH * (int)HW2, Cm, Ccat * 9, Ccat, H2, H2, Cm, 0, 1);
        float* out = (l & 1) ? P1 : P0;
        launch_gemm(MODE_CONV3, mid, oc2w[l], oc2b[l], out,
                    BATCH * (int)HW2, CoC, Cm * 9, Cm, H2, H2, CoC, 0, 0);
        xo = out; oriFinal = out;
    }
    l2norm2_kernel<<<blocks1d((long)BATCH * 262144), TPB, 0, stream>>>(
        oriFinal, o_xo, BATCH, 262144);
}
